// ShiftedWindowAttention_72945724555764
// MI455X (gfx1250) — matrix operand order repaired and checked by reading, not compiled, not run on hardware
//
#include <hip/hip_runtime.h>
#include <hip/hip_bf16.h>
#include <cstdint>

typedef __attribute__((ext_vector_type(16))) _Float16 v16h;
typedef __attribute__((ext_vector_type(8)))  float    v8f;

// ---------------------------------------------------------------------------
// WMMA + async-copy helpers (CDNA5 gfx1250, wave32)
// ---------------------------------------------------------------------------
__device__ inline v8f wmma_f16(v16h a, v16h b, v8f c) {
  return __builtin_amdgcn_wmma_f32_16x16x32_f16(false, a, false, b, (short)0, c,
                                                false, false);
}

// LDS byte offset of a __shared__ object (low 32 bits of generic pointer).
__device__ inline unsigned lds_off(const void* p) {
  return (unsigned)(unsigned long long)(uintptr_t)p;
}

// Async 16-byte copy global memory -> LDS (GLOBAL_LOAD_ASYNC_TO_LDS_B128,
// tracked by ASYNCcnt; completes in order per wave).
__device__ inline void async_copy_b128(unsigned ldsoff, const void* src) {
  asm volatile("global_load_async_to_lds_b128 %0, %1, off"
               :
               : "v"(ldsoff), "v"(src)
               : "memory");
}

template <int N>
__device__ inline void wait_asynccnt() {
  asm volatile("s_wait_asynccnt %0" : : "i"(N) : "memory");
}

// A-matrix 16x32 f16 fragment from row-major LDS (stride in elements).
__device__ inline v16h load_frag_A(const _Float16* base, int stride, int row0,
                                   int lane) {
  v16h a;
  int row = row0 + (lane & 15);
  int kb = (lane < 16) ? 0 : 16;
  const _Float16* p = base + row * stride;
#pragma unroll
  for (int i = 0; i < 8; ++i) {
    int k = kb + 2 * i;
    a[2 * i]     = p[k];
    a[2 * i + 1] = p[k + 1];
  }
  return a;
}

// B-matrix 32x16 where B[k][n] = S[n][k], S row-major [n][stride].
__device__ inline v16h load_frag_B_t(const _Float16* base, int stride, int col0,
                                     int lane) {
  v16h b;
  int col = col0 + (lane & 15);
  int kb = (lane < 16) ? 0 : 16;
  const _Float16* p = base + col * stride + kb;
#pragma unroll
  for (int i = 0; i < 8; ++i) {
    b[2 * i]     = p[2 * i];
    b[2 * i + 1] = p[2 * i + 1];
  }
  return b;
}

// B-matrix 32x16 where B[k][n] = M[k][n], M row-major [k][stride].
__device__ inline v16h load_frag_B_n(const _Float16* base, int stride, int k0,
                                     int col0, int lane) {
  v16h b;
  int col = col0 + (lane & 15);
  int kb = k0 + ((lane < 16) ? 0 : 16);
#pragma unroll
  for (int i = 0; i < 8; ++i) {
    b[2 * i]     = base[(kb + 2 * i) * stride + col];
    b[2 * i + 1] = base[(kb + 2 * i + 1) * stride + col];
  }
  return b;
}

// ---------------------------------------------------------------------------
// Kernel 1a: convert weights f32 -> f16
// ---------------------------------------------------------------------------
__global__ __launch_bounds__(256) void cvt_wt_kernel(
    const float* __restrict__ qkv_w, const float* __restrict__ proj_w,
    _Float16* __restrict__ wq, _Float16* __restrict__ wp) {
  int i = blockIdx.x * 256 + threadIdx.x;
  if (i < 768 * 256) wq[i] = (_Float16)qkv_w[i];
  if (i < 256 * 256) wp[i] = (_Float16)proj_w[i];
}

// Kernel 1b: convert x f32 -> f16 (4 elements per thread, vectorized)
__global__ __launch_bounds__(256) void cvt_x_kernel(const float* __restrict__ x,
                                                    _Float16* __restrict__ xh) {
  size_t i = ((size_t)blockIdx.x * 256 + threadIdx.x) * 4;
  float4 v = *(const float4*)(x + i);
  union { _Float16 h[4]; uint2 u; } pk;
  pk.h[0] = (_Float16)v.x;
  pk.h[1] = (_Float16)v.y;
  pk.h[2] = (_Float16)v.z;
  pk.h[3] = (_Float16)v.w;
  *(uint2*)(xh + i) = pk.u;
}

// ---------------------------------------------------------------------------
// Kernel 2: CPB MLP -> relative position bias rpb[8][64][64] (f32)
// ---------------------------------------------------------------------------
__global__ __launch_bounds__(256) void rpb_kernel(
    const float* __restrict__ w1, const float* __restrict__ b1,
    const float* __restrict__ w2, float* __restrict__ rpb) {
  __shared__ float tbl[225 * 8];
  int t = threadIdx.x;
  if (t < 225) {
    int di = t / 15, dj = t % 15;
    float v0 = (float)(di - 7) * (8.0f / 7.0f);
    float v1 = (float)(dj - 7) * (8.0f / 7.0f);
    float c0 = copysignf(log2f(fabsf(v0) + 1.0f) * (1.0f / 3.0f), v0);
    float c1 = copysignf(log2f(fabsf(v1) + 1.0f) * (1.0f / 3.0f), v1);
    float o[8] = {0.f, 0.f, 0.f, 0.f, 0.f, 0.f, 0.f, 0.f};
    for (int j = 0; j < 512; ++j) {
      float h = fmaxf(c0 * w1[2 * j] + c1 * w1[2 * j + 1] + b1[j], 0.0f);
#pragma unroll
      for (int hd = 0; hd < 8; ++hd) o[hd] += h * w2[hd * 512 + j];
    }
#pragma unroll
    for (int hd = 0; hd < 8; ++hd) tbl[t * 8 + hd] = o[hd];
  }
  __syncthreads();
  for (int e = t; e < 8 * 64 * 64; e += 256) {
    int head = e >> 12;
    int r = (e >> 6) & 63;
    int c = e & 63;
    int di = (r >> 3) - (c >> 3) + 7;
    int dj = (r & 7) - (c & 7) + 7;
    float bv = tbl[(di * 15 + dj) * 8 + head];
    rpb[e] = 16.0f / (1.0f + __expf(-bv));
  }
}

// ---------------------------------------------------------------------------
// Kernel 3: QKV projection fused with cyclic shift + window partition.
// M=64 tokens x N=256 outputs (segment = blockIdx.y), K=256 in 32-chunks.
// Double-buffered LDS fed by GLOBAL_LOAD_ASYNC_TO_LDS_B128 (10 async b128
// per thread per chunk), overlapped with the 16 WMMAs of the previous chunk.
// ---------------------------------------------------------------------------
__global__ __launch_bounds__(128) void qkv_kernel(
    const _Float16* __restrict__ xh, const _Float16* __restrict__ wq,
    const float* __restrict__ qkv_b, _Float16* __restrict__ qh,
    _Float16* __restrict__ kh, _Float16* __restrict__ vh) {
  __shared__ _Float16 xs[2][64 * 32];    // 2 x 4 KB
  __shared__ _Float16 wsm[2][256 * 32];  // 2 x 16 KB

  int win = blockIdx.x;
  int seg = blockIdx.y;  // 0=q 1=k 2=v
  int n0 = seg * 256;
  int t = threadIdx.x, lane = t & 31, wave = t >> 5;

  int b = win >> 6, w6 = win & 63, wh = w6 >> 3, ww = w6 & 7;
  int rq = t >> 2;          // 0..31
  int off8 = (t & 3) * 8;   // element offset of this thread's 16B chunk

  auto xrowbase = [&](int row) -> const _Float16* {
    int i = row >> 3, j = row & 7;
    int h = (wh * 8 + i + 4) & 63;  // roll(x,-4): xs[h'] = x[(h'+4)%64]
    int w = (ww * 8 + j + 4) & 63;
    return xh + (size_t)((b * 64 + h) * 64 + w) * 256;
  };
  const _Float16* xsrcA = xrowbase(rq) + off8;
  const _Float16* xsrcB = xrowbase(rq + 32) + off8;
  const _Float16* wsrc = wq + (size_t)(n0 + rq) * 256 + off8;

  unsigned xdstA = lds_off(&xs[0][rq * 32 + off8]);
  unsigned xdstB = lds_off(&xs[0][(rq + 32) * 32 + off8]);
  unsigned wdst  = lds_off(&wsm[0][rq * 32 + off8]);

  auto issue = [&](int buf, int k0) {
    async_copy_b128(xdstA + buf * 4096, xsrcA + k0);
    async_copy_b128(xdstB + buf * 4096, xsrcB + k0);
#pragma unroll
    for (int j = 0; j < 8; ++j)  // 8 x 32 weight rows
      async_copy_b128(wdst + buf * 16384 + j * 2048, wsrc + k0 + j * 8192);
  };

  v8f acc[16] = {};
  issue(0, 0);
  for (int kk = 0; kk < 8; ++kk) {
    int cur = kk & 1;
    if (kk < 7) {
      issue(1 - cur, 32 * (kk + 1));
      wait_asynccnt<10>();  // in-order: current buffer's 10 copies done
    } else {
      wait_asynccnt<0>();
    }
    __syncthreads();
    v16h a = load_frag_A(&xs[cur][0], 32, wave * 16, lane);
#pragma unroll
    for (int n = 0; n < 16; ++n) {
      v16h bf = load_frag_B_t(&wsm[cur][0], 32, n * 16, lane);
      acc[n] = wmma_f16(a, bf, acc[n]);
    }
    __syncthreads();  // all waves done reading before buffer reuse
  }

  _Float16* dst = (seg == 0) ? qh : (seg == 1) ? kh : vh;
  int colL = lane & 15;
  int rbase = wave * 16 + ((lane < 16) ? 0 : 8);
#pragma unroll
  for (int n = 0; n < 16; ++n) {
    int cc = n * 16 + colL;  // 0..255 within segment
    float bias = qkv_b[n0 + cc];
    int head = cc >> 5, d = cc & 31;
#pragma unroll
    for (int r = 0; r < 8; ++r) {
      int tok = rbase + r;
      dst[(((size_t)win * 8 + head) * 64 + tok) * 32 + d] =
          (_Float16)(acc[n][r] + bias);
    }
  }
}

// ---------------------------------------------------------------------------
// Kernel 4: attention for one (window, head). 128 threads = 4 waves.
// ---------------------------------------------------------------------------
__global__ __launch_bounds__(128) void attn_kernel(
    const _Float16* __restrict__ qh, const _Float16* __restrict__ kh,
    const _Float16* __restrict__ vh, const float* __restrict__ rpb,
    const float* __restrict__ logit_scale, _Float16* __restrict__ ao) {
  __shared__ _Float16 qs[64 * 32], ks[64 * 32], vs[64 * 32];
  __shared__ float sArr[64 * 64];
  __shared__ _Float16 pArr[64 * 64];

  int win = blockIdx.x, head = blockIdx.y;
  int t = threadIdx.x, lane = t & 31, wave = t >> 5;

  size_t base = ((size_t)win * 8 + head) * 64 * 32;
  // 256 x 16B chunks per matrix; 2 per thread, via async copies.
  {
    unsigned lq = lds_off(qs), lk = lds_off(ks), lv = lds_off(vs);
    const _Float16* qp = qh + base;
    const _Float16* kp = kh + base;
    const _Float16* vp = vh + base;
#pragma unroll
    for (int it = 0; it < 2; ++it) {
      int c = t + 128 * it;
      async_copy_b128(lq + c * 16, qp + c * 8);
      async_copy_b128(lk + c * 16, kp + c * 8);
      async_copy_b128(lv + c * 16, vp + c * 8);
    }
    wait_asynccnt<0>();
  }
  __syncthreads();

  // row-wise cosine normalization; logit scale folded into q
  {
    float scale = __expf(fminf(logit_scale[head], 4.60517019f));  // ln(100)
    int row = t & 63;
    _Float16* p = (t < 64) ? (qs + row * 32) : (ks + row * 32);
    float ss = 0.0f;
#pragma unroll
    for (int i = 0; i < 32; ++i) {
      float v = (float)p[i];
      ss += v * v;
    }
    float sc = 1.0f / fmaxf(sqrtf(ss), 1e-12f);
    if (t < 64) sc *= scale;
#pragma unroll
    for (int i = 0; i < 32; ++i) p[i] = (_Float16)((float)p[i] * sc);
  }
  __syncthreads();

  // S = qn @ kn^T : wave w owns rows [16w,16w+16), 4 WMMAs (K=32).
  {
    v16h a = load_frag_A(qs, 32, wave * 16, lane);
#pragma unroll
    for (int n = 0; n < 4; ++n) {
      v16h b = load_frag_B_t(ks, 32, n * 16, lane);
      v8f c = {};
      c = wmma_f16(a, b, c);
      int col = n * 16 + (lane & 15);
      int r0 = wave * 16 + ((lane < 16) ? 0 : 8);
#pragma unroll
      for (int r = 0; r < 8; ++r) sArr[(r0 + r) * 64 + col] = c[r];
    }
  }
  __syncthreads();

  // softmax: 2 threads per row (32 cols each), pair-combined via shfl_xor.
  {
    int r = t >> 1;
    int cbase = (t & 1) * 32;
    int w6 = win & 63, wh = w6 >> 3, ww = w6 & 7;
    int ir = r >> 3, jr = r & 7;
    int hr = wh * 8 + ir, wr2 = ww * 8 + jr;
    int idr = ((hr < 56) ? 0 : (hr < 60) ? 1 : 2) * 3 +
              ((wr2 < 56) ? 0 : (wr2 < 60) ? 1 : 2);
    const float* rp = rpb + (head * 64 + r) * 64;
    float mx = -1e30f;
#pragma unroll
    for (int c2 = 0; c2 < 32; ++c2) {
      int cc = cbase + c2;
      int ic = cc >> 3, jc = cc & 7;
      int hc = wh * 8 + ic, wc = ww * 8 + jc;
      int idc = ((hc < 56) ? 0 : (hc < 60) ? 1 : 2) * 3 +
                ((wc < 56) ? 0 : (wc < 60) ? 1 : 2);
      float s = sArr[r * 64 + cc] + rp[cc];
      if (idc != idr) s -= 100.0f;
      sArr[r * 64 + cc] = s;
      mx = fmaxf(mx, s);
    }
    mx = fmaxf(mx, __shfl_xor(mx, 1));
    float sum = 0.0f;
#pragma unroll
    for (int c2 = 0; c2 < 32; ++c2) {
      int cc = cbase + c2;
      float e = __expf(sArr[r * 64 + cc] - mx);
      sArr[r * 64 + cc] = e;
      sum += e;
    }
    sum += __shfl_xor(sum, 1);
    float inv = 1.0f / sum;
#pragma unroll
    for (int c2 = 0; c2 < 32; ++c2) {
      int cc = cbase + c2;
      pArr[r * 64 + cc] = (_Float16)(sArr[r * 64 + cc] * inv);
    }
  }
  __syncthreads();

  // P = attn @ v : 4x2 tiles, K=64 (2 WMMAs each)
#pragma unroll
  for (int n = 0; n < 2; ++n) {
    v8f c = {};
#pragma unroll
    for (int kh2 = 0; kh2 < 2; ++kh2) {
      v16h a = load_frag_A(pArr + kh2 * 32, 64, wave * 16, lane);
      v16h b = load_frag_B_n(vs, 32, kh2 * 32, n * 16, lane);
      c = wmma_f16(a, b, c);
    }
    int col = n * 16 + (lane & 15);
    int r0 = wave * 16 + ((lane < 16) ? 0 : 8);
#pragma unroll
    for (int r = 0; r < 8; ++r) {
      int tok = r0 + r;
      ao[((size_t)win * 64 + tok) * 256 + head * 32 + col] = (_Float16)c[r];
    }
  }
}

// ---------------------------------------------------------------------------
// Kernel 5: output projection + window merge + reverse cyclic shift.
// Same async double-buffered pipeline as kernel 3 (input already f16).
// ---------------------------------------------------------------------------
__global__ __launch_bounds__(128) void proj_kernel(
    const _Float16* __restrict__ ao, const _Float16* __restrict__ wp,
    const float* __restrict__ proj_b, float* __restrict__ out) {
  __shared__ _Float16 xs[2][64 * 32];
  __shared__ _Float16 wsm[2][256 * 32];

  int win = blockIdx.x;
  int t = threadIdx.x, lane = t & 31, wave = t >> 5;

  int rq = t >> 2;
  int off8 = (t & 3) * 8;

  const _Float16* xsrcA = ao + (size_t)win * 64 * 256 + rq * 256 + off8;
  const _Float16* xsrcB = xsrcA + 32 * 256;
  const _Float16* wsrc = wp + (size_t)rq * 256 + off8;

  unsigned xdstA = lds_off(&xs[0][rq * 32 + off8]);
  unsigned xdstB = lds_off(&xs[0][(rq + 32) * 32 + off8]);
  unsigned wdst  = lds_off(&wsm[0][rq * 32 + off8]);

  auto issue = [&](int buf, int k0) {
    async_copy_b128(xdstA + buf * 4096, xsrcA + k0);
    async_copy_b128(xdstB + buf * 4096, xsrcB + k0);
#pragma unroll
    for (int j = 0; j < 8; ++j)
      async_copy_b128(wdst + buf * 16384 + j * 2048, wsrc + k0 + j * 8192);
  };

  v8f acc[16] = {};
  issue(0, 0);
  for (int kk = 0; kk < 8; ++kk) {
    int cur = kk & 1;
    if (kk < 7) {
      issue(1 - cur, 32 * (kk + 1));
      wait_asynccnt<10>();
    } else {
      wait_asynccnt<0>();
    }
    __syncthreads();
    v16h a = load_frag_A(&xs[cur][0], 32, wave * 16, lane);
#pragma unroll
    for (int n = 0; n < 16; ++n) {
      v16h bf = load_frag_B_t(&wsm[cur][0], 32, n * 16, lane);
      acc[n] = wmma_f16(a, bf, acc[n]);
    }
    __syncthreads();
  }

  int b = win >> 6, w6 = win & 63, wh = w6 >> 3, ww = w6 & 7;
  int rbase = wave * 16 + ((lane < 16) ? 0 : 8);
#pragma unroll
  for (int n = 0; n < 16; ++n) {
    int col = n * 16 + (lane & 15);
    float bias = proj_b[col];
#pragma unroll
    for (int r = 0; r < 8; ++r) {
      int tok = rbase + r;
      int i = tok >> 3, j = tok & 7;
      int h = (wh * 8 + i + 4) & 63;  // reverse shift
      int w = (ww * 8 + j + 4) & 63;
      out[(((size_t)b * 64 + h) * 64 + w) * 256 + col] = acc[n][r] + bias;
    }
  }
}

// ---------------------------------------------------------------------------
// Host launcher
// ---------------------------------------------------------------------------
extern "C" void kernel_launch(void* const* d_in, const int* in_sizes, int n_in,
                              void* d_out, int out_size, void* d_ws,
                              size_t ws_size, hipStream_t stream) {
  const float* x      = (const float*)d_in[0];
  const float* qkv_w  = (const float*)d_in[1];
  const float* qkv_b  = (const float*)d_in[2];
  const float* proj_w = (const float*)d_in[3];
  const float* proj_b = (const float*)d_in[4];
  const float* lscale = (const float*)d_in[5];
  const float* cpb_w1 = (const float*)d_in[6];
  const float* cpb_b1 = (const float*)d_in[7];
  const float* cpb_w2 = (const float*)d_in[8];
  float* out = (float*)d_out;

  char* ws = (char*)d_ws;
  size_t off = 0;
  auto take = [&](size_t bytes) {
    size_t r = off;
    off = (off + bytes + 255) & ~(size_t)255;
    return r;
  };
  _Float16* wq_h = (_Float16*)(ws + take(768 * 256 * 2));
  _Float16* wp_h = (_Float16*)(ws + take(256 * 256 * 2));
  float* rpb     = (float*)(ws + take(8 * 64 * 64 * 4));
  const size_t qkv_elems = (size_t)2048 * 8 * 64 * 32;  // per matrix
  _Float16* q_h  = (_Float16*)(ws + take(qkv_elems * 2));
  _Float16* k_h  = (_Float16*)(ws + take(qkv_elems * 2));
  _Float16* v_h  = (_Float16*)(ws + take(qkv_elems * 2));
  _Float16* x_h  = (_Float16*)(ws + take((size_t)32 * 64 * 64 * 256 * 2));
  _Float16* ao_h = x_h;  // x_h is dead after qkv_kernel; alias for attn output
  (void)ws_size; (void)in_sizes; (void)n_in; (void)out_size;

  cvt_wt_kernel<<<dim3(768), dim3(256), 0, stream>>>(qkv_w, proj_w, wq_h, wp_h);
  cvt_x_kernel<<<dim3(32768), dim3(256), 0, stream>>>(x, x_h);
  rpb_kernel<<<dim3(1), dim3(256), 0, stream>>>(cpb_w1, cpb_b1, cpb_w2, rpb);
  qkv_kernel<<<dim3(2048, 3), dim3(128), 0, stream>>>(x_h, wq_h, qkv_b, q_h,
                                                      k_h, v_h);
  attn_kernel<<<dim3(2048, 8), dim3(128), 0, stream>>>(q_h, k_h, v_h, rpb,
                                                       lscale, ao_h);
  proj_kernel<<<dim3(2048), dim3(128), 0, stream>>>(ao_h, wp_h, proj_b, out);
}
